// SparseConvNet_18614388261288
// MI455X (gfx1250) — matrix operand order repaired and checked by reading, not compiled, not run on hardware
//
#include <hip/hip_runtime.h>
#include <stdint.h>

// ---------------------------------------------------------------------------
// SparseConvNet for MI455X (gfx1250): implicit-GEMM partial convolutions using
// V_WMMA_F32_16X16X32_F16, vertically tiled (8 output rows / workgroup) so the
// (k x patch) activation reads amortize. Activations ping-pong as f16 NHWC
// (128 MB, fits in the 192 MB L2). Weights staged into LDS with
// GLOBAL_LOAD_ASYNC_TO_LDS_B128; results drained with
// GLOBAL_STORE_ASYNC_FROM_LDS_B128 (both on ASYNCcnt).
// ---------------------------------------------------------------------------

typedef _Float16 half_t;
typedef __attribute__((ext_vector_type(16))) _Float16 v16h;
typedef __attribute__((ext_vector_type(8)))  _Float16 v8h;
typedef __attribute__((ext_vector_type(8)))  float    v8f;

#define NB 4
#define HH 1024
#define WW 1024
#define HW (HH * WW)
#define TW 128                 // output pixels per workgroup row tile
#define TH 8                   // output rows per workgroup
#define COUNT_F 4194304.0f     // N*H*W elements per channel for BN

__device__ __forceinline__ v8h v8h_zero() {
  v8h r;
#pragma unroll
  for (int i = 0; i < 8; ++i) r[i] = (half_t)0.0f;
  return r;
}
__device__ __forceinline__ v8f v8f_zero() {
  v8f r;
#pragma unroll
  for (int i = 0; i < 8; ++i) r[i] = 0.0f;
  return r;
}

// Async DMA: global <-> LDS, 16 bytes per issue, tracked by ASYNCcnt.
__device__ __forceinline__ void async_ld_b128(unsigned lds_off, const void* gptr) {
  asm volatile("global_load_async_to_lds_b128 %0, %1, off"
               :: "v"(lds_off), "v"((uint64_t)(uintptr_t)gptr) : "memory");
}
__device__ __forceinline__ void async_st_b128(void* gptr, unsigned lds_off) {
  asm volatile("global_store_async_from_lds_b128 %0, %1, off"
               :: "v"((uint64_t)(uintptr_t)gptr), "v"(lds_off) : "memory");
}
__device__ __forceinline__ void wait_async0() {
  asm volatile("s_wait_asynccnt 0" ::: "memory");
}

// ---------------------------------------------------------------------------
// small utility kernels
// ---------------------------------------------------------------------------
__global__ void zero_kernel(float* p, int n) {
  int i = blockIdx.x * blockDim.x + threadIdx.x;
  if (i < n) p[i] = 0.0f;
}

// w1 (16,1,11,11) -> f16 [cout][128] (taps 121..127 zero padded)
__global__ void repack_w1_kernel(const float* __restrict__ w, half_t* __restrict__ out) {
  int i = blockIdx.x * blockDim.x + threadIdx.x;   // 16*128
  if (i < 16 * 128) {
    int c = i >> 7, t = i & 127;
    out[i] = (half_t)((t < 121) ? w[c * 121 + t] : 0.0f);
  }
}

// w (16,16,k,k) OIHW -> f16 [tap][cout][cin]
__global__ void repack_w_kernel(const float* __restrict__ w, half_t* __restrict__ out, int k) {
  int i = blockIdx.x * blockDim.x + threadIdx.x;
  int kk = k * k;
  if (i < kk * 256) {
    int t = i >> 8, o = (i >> 4) & 15, ci = i & 15;
    out[(t * 16 + o) * 16 + ci] = (half_t)w[(o * 16 + ci) * kk + t];
  }
}

// finalize BN: scale = g*rsqrt(var+eps), shift = beta - mean*scale
__global__ void bn_stats_kernel(const float* __restrict__ stats, const float* __restrict__ g,
                                const float* __restrict__ beta, float* __restrict__ scales, int nc) {
  int c = threadIdx.x;
  if (c < nc) {
    float mean = stats[c] * (1.0f / COUNT_F);
    float var  = stats[16 + c] * (1.0f / COUNT_F) - mean * mean;
    float rstd = rsqrtf(var + 1e-5f);
    float sc   = g[c] * rstd;
    scales[c]      = sc;
    scales[16 + c] = beta[c] - mean * sc;
  }
}

// ---------------------------------------------------------------------------
// Layer 1: cin=1, cout=16, k=11.  Implicit GEMM, K = 121 taps padded to 128.
// Tap -> LDS-offset mapping resolved at compile time; invalid taps hit zeroed
// rows 18..25 (valid for any row base r*PW, r<8).
// ---------------------------------------------------------------------------
#define C1_PW (TW + 10)          // 138
#define C1_PR (TH + 10)          // 18 real patch rows
__device__ constexpr int tap_off(int t) {
  return (t < 121) ? (t / 11) * C1_PW + (t % 11) : C1_PR * C1_PW;
}

__global__ __launch_bounds__(256) void conv1_kernel(
    const float* __restrict__ x, const half_t* __restrict__ w1p,
    const float* __restrict__ bias, half_t* __restrict__ actOut,
    float* __restrict__ maskOut, float* __restrict__ stats) {
  constexpr int KS = 11, P = 5, PW = C1_PW, PR = C1_PR;

  __shared__ alignas(16) half_t lpatch[PR + TH][PW];   // rows PR..PR+7 zeroed
  __shared__ float  lmask[PR][PW];
  __shared__ alignas(16) half_t lw[16 * 128];
  __shared__ alignas(16) half_t dtile[TW][16];
  __shared__ float  lstat[32];

  const int tid = threadIdx.x;
  const int bid = blockIdx.x;
  const int tx  = bid & 7;
  const int ry  = (bid >> 3) & 127;
  const int nn  = bid >> 10;
  const int x0  = tx * TW;
  const int y0  = ry * TH;

  if (tid < 32) lstat[tid] = 0.0f;

  // async DMA the repacked weights straight into LDS (4096 B = 256 x b128)
  {
    unsigned lbase = (unsigned)(uintptr_t)(void*)&lw[0];
    async_ld_b128(lbase + tid * 16, (const char*)w1p + tid * 16);
  }

  for (int i = tid; i < (PR + TH) * PW; i += 256) {
    int ky = i / PW, px = i % PW;
    int gy = y0 - P + ky, gx = x0 - P + px;
    float d = 0.0f, m = 0.0f;
    if (ky < PR && gy >= 0 && gy < HH && gx >= 0 && gx < WW) {
      size_t base = (size_t)gy * WW + gx;
      d = x[(size_t)(nn * 2) * HW + base];
      m = x[(size_t)(nn * 2 + 1) * HW + base];
    }
    if (ky < PR) lmask[ky][px] = m;
    lpatch[ky][px] = (half_t)(d * m);
  }
  wait_async0();
  __syncthreads();

  const int wave = tid >> 5, lane = tid & 31;
  const int laneM = lane & 15, laneHi = lane >> 4;
  const int tb0 = wave * 16;
  const int colbase = tb0 + laneM;
  const half_t* lpflat = &lpatch[0][0];

  // separable mask window stats: per-patch-row sums and maxes for this pixel
  float rowS[PR], rowM[PR];
#pragma unroll
  for (int ryi = 0; ryi < PR; ++ryi) {
    float s = 0.0f, mx = 0.0f;
#pragma unroll
    for (int kx = 0; kx < KS; ++kx) {
      float m = lmask[ryi][colbase + kx];
      s += m;
      mx = fmaxf(mx, m);
    }
    rowS[ryi] = s;
    rowM[ryi] = mx;
  }

  const float bc = bias[laneM];
  float psum = 0.0f, psq = 0.0f;

  for (int r = 0; r < TH; ++r) {
    const int rb = colbase + r * PW;
    v8f acc = v8f_zero();
#pragma unroll
    for (int T = 0; T < 128; T += 32) {
      union { v16h v; half_t e[16]; } A;
      union { v16h v; v8h h[2]; } B;
#pragma unroll
      for (int v = 0; v < 8; ++v) {
        const int K0  = (v < 4 ? 2 * v : 16 + 2 * (v - 4));
        const int tA0 = T + K0;        // taps when laneHi == 0
        const int tB0 = T + K0 + 8;    // taps when laneHi == 1
        int off0 = laneHi ? tap_off(tB0)     : tap_off(tA0);
        int off1 = laneHi ? tap_off(tB0 + 1) : tap_off(tA0 + 1);
        A.e[2 * v]     = lpflat[rb + off0];
        A.e[2 * v + 1] = lpflat[rb + off1];
      }
      const half_t* bp = &lw[laneM * 128 + T + laneHi * 8];
      B.h[0] = *(const v8h*)bp;
      B.h[1] = *(const v8h*)(bp + 16);
      acc = __builtin_amdgcn_wmma_f32_16x16x32_f16(false, A.v, false, B.v, (short)0, acc, false, false);
    }

    float ssum = 0.0f, mmax = 0.0f;
#pragma unroll
    for (int ky = 0; ky < KS; ++ky) {
      ssum += rowS[r + ky];
      mmax = fmaxf(mmax, rowM[r + ky]);
    }
    float recip = 1.0f / (ssum + 1e-8f);
    if (laneHi == 0)
      maskOut[(size_t)(nn * HH + y0 + r) * WW + x0 + colbase] = mmax;

#pragma unroll
    for (int rr = 0; rr < 8; ++rr) {
      int q = rr + laneHi * 8;
      float rp = __shfl(recip, q, 32);
      float yv = acc[rr] * rp + bc;
      psum += yv;
      psq  += yv * yv;
      dtile[tb0 + q][laneM] = (half_t)yv;
    }
    __syncthreads();
    {
      unsigned loff = (unsigned)(uintptr_t)(void*)&dtile[0][0] + tid * 16;
      int p = tid >> 1, hsel = tid & 1;
      async_st_b128(actOut + ((size_t)(nn * HH + y0 + r) * WW + x0 + p) * 16 + hsel * 8, loff);
    }
    wait_async0();
    __syncthreads();
  }

  atomicAdd(&lstat[laneM], psum);
  atomicAdd(&lstat[16 + laneM], psq);
  __syncthreads();
  if (tid < 32) atomicAdd(&stats[tid], lstat[tid]);
}

// ---------------------------------------------------------------------------
// Layers 2..5: cin=16, cout=16, k in {7,5,3}. One WMMA per pair of taps,
// TH output rows per workgroup. Applies previous layer's BN + ReLU + mask
// while staging the patch.
// ---------------------------------------------------------------------------
template <int KS>
__global__ __launch_bounds__(256) void conv16_kernel(
    const half_t* __restrict__ actIn, const float* __restrict__ scaleShift,
    const float* __restrict__ maskIn, const half_t* __restrict__ wPack,
    const float* __restrict__ bias, half_t* __restrict__ actOut,
    float* __restrict__ maskOut, float* __restrict__ stats) {
  constexpr int P = KS / 2;
  constexpr int PW = TW + KS - 1;
  constexpr int PR = TH + KS - 1;
  constexpr int NT = KS * KS;

  __shared__ alignas(16) half_t lpatch[PR][PW][16];   // normalized*mask, f16
  __shared__ float  lmask[PR][PW];
  __shared__ alignas(16) half_t lw[NT * 256];
  __shared__ alignas(16) half_t lzero[16];
  __shared__ alignas(16) half_t dtile[TW][16];
  __shared__ float  lstat[32];

  const int tid = threadIdx.x;
  const int bid = blockIdx.x;
  const int tx  = bid & 7;
  const int ry  = (bid >> 3) & 127;
  const int nn  = bid >> 10;
  const int x0  = tx * TW;
  const int y0  = ry * TH;

  float sc[16], sh[16];
#pragma unroll
  for (int c = 0; c < 16; ++c) { sc[c] = scaleShift[c]; sh[c] = scaleShift[16 + c]; }

  if (tid < 16) lzero[tid] = (half_t)0.0f;
  if (tid < 32) lstat[tid] = 0.0f;

  // async DMA the repacked weights straight into LDS (NT*512 B = NT*32 x b128)
  {
    unsigned lbase = (unsigned)(uintptr_t)(void*)&lw[0];
    for (int i = tid; i < NT * 32; i += 256)
      async_ld_b128(lbase + i * 16, (const char*)wPack + i * 16);
  }

  for (int i = tid; i < PR * PW; i += 256) {
    int ky = i / PW, px = i % PW;
    int gy = y0 - P + ky, gx = x0 - P + px;
    float m = 0.0f;
    v8h o0 = v8h_zero(), o1 = v8h_zero();
    if (gy >= 0 && gy < HH && gx >= 0 && gx < WW) {
      m = maskIn[(size_t)(nn * HH + gy) * WW + gx];
      const v8h* ap = (const v8h*)(actIn + ((size_t)(nn * HH + gy) * WW + gx) * 16);
      v8h a0 = ap[0], a1 = ap[1];
#pragma unroll
      for (int c = 0; c < 8; ++c) o0[c] = (half_t)(fmaxf(0.0f, (float)a0[c] * sc[c] + sh[c]) * m);
#pragma unroll
      for (int c = 0; c < 8; ++c) o1[c] = (half_t)(fmaxf(0.0f, (float)a1[c] * sc[c + 8] + sh[c + 8]) * m);
    }
    lmask[ky][px] = m;
    *(v8h*)&lpatch[ky][px][0] = o0;
    *(v8h*)&lpatch[ky][px][8] = o1;
  }
  wait_async0();
  __syncthreads();

  const int wave = tid >> 5, lane = tid & 31;
  const int laneM = lane & 15, laneHi = lane >> 4;
  const int tb0 = wave * 16;
  const int colbase = tb0 + laneM;

  // separable mask window stats
  float rowS[PR], rowM[PR];
#pragma unroll
  for (int ryi = 0; ryi < PR; ++ryi) {
    float s = 0.0f, mx = 0.0f;
#pragma unroll
    for (int kx = 0; kx < KS; ++kx) {
      float m = lmask[ryi][colbase + kx];
      s += m;
      mx = fmaxf(mx, m);
    }
    rowS[ryi] = s;
    rowM[ryi] = mx;
  }

  const float bc = bias[laneM];
  float psum = 0.0f, psq = 0.0f;

  for (int r = 0; r < TH; ++r) {
    v8f acc = v8f_zero();
#pragma unroll
    for (int t = 0; t < NT; t += 2) {
      const int ta = t, tb = t + 1;
      union { v16h v; v8h h[2]; } A, B;
      {
        int kyA = ta / KS, kxA = ta % KS;
        A.h[0] = *(const v8h*)&lpatch[r + kyA][colbase + kxA][laneHi * 8];
      }
      if (tb < NT) {
        int kyB = tb / KS, kxB = tb % KS;
        A.h[1] = *(const v8h*)&lpatch[r + kyB][colbase + kxB][laneHi * 8];
      } else {
        A.h[1] = *(const v8h*)&lzero[0];
      }
      B.h[0] = *(const v8h*)&lw[(ta * 16 + laneM) * 16 + laneHi * 8];
      if (tb < NT) {
        B.h[1] = *(const v8h*)&lw[(tb * 16 + laneM) * 16 + laneHi * 8];
      } else {
        B.h[1] = *(const v8h*)&lzero[0];
      }
      acc = __builtin_amdgcn_wmma_f32_16x16x32_f16(false, A.v, false, B.v, (short)0, acc, false, false);
    }

    float ssum = 0.0f, mmax = 0.0f;
#pragma unroll
    for (int ky = 0; ky < KS; ++ky) {
      ssum += rowS[r + ky];
      mmax = fmaxf(mmax, rowM[r + ky]);
    }
    float recip = 1.0f / (ssum + 1e-8f);
    if (laneHi == 0)
      maskOut[(size_t)(nn * HH + y0 + r) * WW + x0 + colbase] = mmax;

#pragma unroll
    for (int rr = 0; rr < 8; ++rr) {
      int q = rr + laneHi * 8;
      float rp = __shfl(recip, q, 32);
      float yv = acc[rr] * rp + bc;
      psum += yv;
      psq  += yv * yv;
      dtile[tb0 + q][laneM] = (half_t)yv;
    }
    __syncthreads();
    {
      unsigned loff = (unsigned)(uintptr_t)(void*)&dtile[0][0] + tid * 16;
      int p = tid >> 1, hsel = tid & 1;
      async_st_b128(actOut + ((size_t)(nn * HH + y0 + r) * WW + x0 + p) * 16 + hsel * 8, loff);
    }
    wait_async0();
    __syncthreads();
  }

  atomicAdd(&lstat[laneM], psum);
  atomicAdd(&lstat[16 + laneM], psq);
  __syncthreads();
  if (tid < 32) atomicAdd(&stats[tid], lstat[tid]);
}

// ---------------------------------------------------------------------------
// Layer 6: 1x1, cin=16, cout=1. y = (sum_c relu(bn(h_c))*w_c) * m/(m+eps) + b
// ---------------------------------------------------------------------------
__global__ __launch_bounds__(256) void conv6_kernel(
    const half_t* __restrict__ actIn, const float* __restrict__ scaleShift,
    const float* __restrict__ maskIn, const float* __restrict__ w6,
    const float* __restrict__ b6, float* __restrict__ t6, float* __restrict__ stats) {
  __shared__ float red[2];
  if (threadIdx.x < 2) red[threadIdx.x] = 0.0f;
  __syncthreads();

  int i = blockIdx.x * 256 + threadIdx.x;
  float t = 0.0f;
  if (i < NB * HW) {
    const v8h* ap = (const v8h*)(actIn + (size_t)i * 16);
    v8h a0 = ap[0], a1 = ap[1];
    float acc = 0.0f;
#pragma unroll
    for (int c = 0; c < 8; ++c)
      acc += fmaxf(0.0f, (float)a0[c] * scaleShift[c] + scaleShift[16 + c]) * w6[c];
#pragma unroll
    for (int c = 0; c < 8; ++c)
      acc += fmaxf(0.0f, (float)a1[c] * scaleShift[c + 8] + scaleShift[16 + c + 8]) * w6[c + 8];
    float m = maskIn[i];
    t = acc * (m / (m + 1e-8f)) + b6[0];
    t6[i] = t;
  }
  atomicAdd(&red[0], t);
  atomicAdd(&red[1], t * t);
  __syncthreads();
  if (threadIdx.x == 0) {
    atomicAdd(&stats[0], red[0]);
    atomicAdd(&stats[16], red[1]);
  }
}

__global__ void final6_kernel(const float* __restrict__ t6, const float* __restrict__ scales,
                              float* __restrict__ out) {
  int i = blockIdx.x * 256 + threadIdx.x;
  if (i < NB * HW) out[i] = t6[i] * scales[0] + scales[16];
}

// ---------------------------------------------------------------------------
extern "C" void kernel_launch(void* const* d_in, const int* in_sizes, int n_in,
                              void* d_out, int out_size, void* d_ws, size_t ws_size,
                              hipStream_t stream) {
  (void)in_sizes; (void)n_in; (void)out_size; (void)ws_size;
  const float* x = (const float*)d_in[0];
  const float *w[6], *b[6], *g[6], *bt[6];
  for (int l = 0; l < 6; ++l) {
    w[l]  = (const float*)d_in[1 + 4 * l];
    b[l]  = (const float*)d_in[2 + 4 * l];
    g[l]  = (const float*)d_in[3 + 4 * l];
    bt[l] = (const float*)d_in[4 + 4 * l];
  }

  char* ws = (char*)d_ws;
  const size_t ACT_BYTES  = (size_t)NB * HW * 16 * 2;   // 128 MB, f16 NHWC
  const size_t MASK_BYTES = (size_t)NB * HW * 4;        // 16 MB, f32
  half_t* act0  = (half_t*)ws;
  half_t* act1  = (half_t*)(ws + ACT_BYTES);
  float*  maskA = (float*)(ws + 2 * ACT_BYTES);
  float*  maskB = (float*)(ws + 2 * ACT_BYTES + MASK_BYTES);
  float*  t6    = (float*)(ws + 2 * ACT_BYTES + 2 * MASK_BYTES);
  float*  stats = (float*)(ws + 2 * ACT_BYTES + 3 * MASK_BYTES);
  float*  scales = stats + 192;                          // 6 layers x {scale16, shift16}
  half_t* w1p = (half_t*)((char*)stats + 4096);
  half_t* w2p = w1p + 16 * 128;
  half_t* w3p = w2p + 49 * 256;
  half_t* w4p = w3p + 25 * 256;
  half_t* w5p = w4p + 9 * 256;

  dim3 blk(256);
  zero_kernel<<<2, blk, 0, stream>>>(stats, 384);
  repack_w1_kernel<<<8, blk, 0, stream>>>(w[0], w1p);
  repack_w_kernel<<<(49 * 256 + 255) / 256, blk, 0, stream>>>(w[1], w2p, 7);
  repack_w_kernel<<<(25 * 256 + 255) / 256, blk, 0, stream>>>(w[2], w3p, 5);
  repack_w_kernel<<<(9 * 256 + 255) / 256, blk, 0, stream>>>(w[3], w4p, 3);
  repack_w_kernel<<<(9 * 256 + 255) / 256, blk, 0, stream>>>(w[4], w5p, 3);

  const int NBLK = NB * (HH / TH) * (WW / TW);   // 4096 workgroups
  conv1_kernel<<<NBLK, blk, 0, stream>>>(x, w1p, b[0], act0, maskA, stats + 0 * 32);
  bn_stats_kernel<<<1, 32, 0, stream>>>(stats + 0 * 32, g[0], bt[0], scales + 0 * 32, 16);
  conv16_kernel<7><<<NBLK, blk, 0, stream>>>(act0, scales + 0 * 32, maskA, w2p, b[1], act1, maskB, stats + 1 * 32);
  bn_stats_kernel<<<1, 32, 0, stream>>>(stats + 1 * 32, g[1], bt[1], scales + 1 * 32, 16);
  conv16_kernel<5><<<NBLK, blk, 0, stream>>>(act1, scales + 1 * 32, maskB, w3p, b[2], act0, maskA, stats + 2 * 32);
  bn_stats_kernel<<<1, 32, 0, stream>>>(stats + 2 * 32, g[2], bt[2], scales + 2 * 32, 16);
  conv16_kernel<3><<<NBLK, blk, 0, stream>>>(act0, scales + 2 * 32, maskA, w4p, b[3], act1, maskB, stats + 3 * 32);
  bn_stats_kernel<<<1, 32, 0, stream>>>(stats + 3 * 32, g[3], bt[3], scales + 3 * 32, 16);
  conv16_kernel<3><<<NBLK, blk, 0, stream>>>(act1, scales + 3 * 32, maskB, w5p, b[4], act0, maskA, stats + 4 * 32);
  bn_stats_kernel<<<1, 32, 0, stream>>>(stats + 4 * 32, g[4], bt[4], scales + 4 * 32, 16);

  const int P6 = (NB * HW + 255) / 256;
  conv6_kernel<<<P6, blk, 0, stream>>>(act0, scales + 4 * 32, maskA, w[5], b[5], t6, stats + 5 * 32);
  bn_stats_kernel<<<1, 32, 0, stream>>>(stats + 5 * 32, g[5], bt[5], scales + 5 * 32, 1);
  final6_kernel<<<P6, blk, 0, stream>>>(t6, scales + 5 * 32, (float*)d_out);
}
